// MSBlock_6614249636698
// MI455X (gfx1250) — matrix operand order repaired and checked by reading, not compiled, not run on hardware
//
#include <hip/hip_runtime.h>

#define B_  128
#define C_  256
#define L_  512
#define NB_ 4
#define BC_ 64

typedef __attribute__((ext_vector_type(16))) __bf16 v16bf;
typedef __attribute__((ext_vector_type(8)))  float  v8f;

union Frag { v16bf v; uint4 q[2]; };

__device__ __forceinline__ unsigned short f2bfu(float f) {
  unsigned u = __float_as_uint(f);
  u += 0x7FFFu + ((u >> 16) & 1u);          // round-to-nearest-even
  return (unsigned short)(u >> 16);
}

__device__ __forceinline__ unsigned pack2bf(float lo, float hi) {
  return (unsigned)f2bfu(lo) | ((unsigned)f2bfu(hi) << 16);
}

__constant__ float c_rec_lo[8] = {
  0.23037781330885523f,  0.7148465705525415f,  0.6308807679295904f,
 -0.02798376941698385f, -0.18703481171888114f, 0.030841381835986965f,
  0.032883011666982945f, -0.010597401784997278f };

// ---------------------------------------------------------------------------
// Generic bf16 WMMA GEMM:  Y[b,m,n] = sum_k Wg[m,k] * X(b,k,n) + bias[m] (+resid)
// X split across two buffers at k == Ksplit (for the concat-input GEMM).
// Optional BN stats (sum / sumsq per channel m) via LDS + global atomics.
// Tile: 64(M) x 128(N) per 256-thread WG; wave w owns n-tile w, 4 m-tiles.
// Software-pipelined: regs <- global(k+1) issued before barrier, WMMA on LDS(k),
// double-buffered LDS -> one barrier per K-step.
// ---------------------------------------------------------------------------
__global__ __launch_bounds__(256) void gemm_bf16(
    const unsigned short* __restrict__ Wg,   // [M,K] bf16
    const float* __restrict__ X,             // [B,Ksplit,L] fp32
    const float* __restrict__ X2,            // [B,K-Ksplit,L] fp32
    const float* __restrict__ bias,          // [M]
    const float* __restrict__ resid,         // [B,M,L] or null
    float* __restrict__ Y,                   // [B,M,L]
    float* __restrict__ stats,               // [2*M] or null
    int M, int K, int L, int Ksplit)
{
  __shared__ __align__(16) unsigned short lds_a[2][64 * 32];
  __shared__ __align__(16) unsigned short lds_x[2][128 * 32];
  __shared__ float sdsum[64];
  __shared__ float sdsq[64];

  const int b    = blockIdx.z;
  const int m0   = blockIdx.y * 64;
  const int n0   = blockIdx.x * 128;
  const int tid  = threadIdx.x;
  const int lane = tid & 31;
  const int wave = tid >> 5;
  const int K2   = K - Ksplit;

  if (tid < 64) { sdsum[tid] = 0.f; sdsq[tid] = 0.f; }

  // per-thread staging coordinates
  const int am  = tid >> 2;            // A row (0..63)
  const int akk = (tid & 3) << 3;      // A k offset {0,8,16,24}
  const int xnn = (tid & 31) << 2;     // X n offset (4 columns)
  const int xkp = (tid >> 5) << 1;     // X k-pair base {0,2,..,14}

  uint4  ra;                           // A chunk in flight
  float4 rx[2][2];                     // X rows (k, k+1) x {low16, high16 k-half}

  auto loadG = [&](int ko) {
    ra = *(const uint4*)&Wg[(size_t)(m0 + am) * K + ko + akk];
#pragma unroll
    for (int r = 0; r < 2; ++r) {
#pragma unroll
      for (int d = 0; d < 2; ++d) {
        int kg = ko + (r << 4) + xkp + d;
        const float* src = (kg < Ksplit)
            ? (X  + ((size_t)b * Ksplit + kg) * L + n0 + xnn)
            : (X2 + ((size_t)b * K2 + (kg - Ksplit)) * L + n0 + xnn);
        rx[r][d] = *(const float4*)src;
      }
    }
  };

  auto storeL = [&](int bi) {
    *(uint4*)&lds_a[bi][am * 32 + akk] = ra;
#pragma unroll
    for (int r = 0; r < 2; ++r) {
      int kb = (r << 4) + xkp;
      float a0[4] = {rx[r][0].x, rx[r][0].y, rx[r][0].z, rx[r][0].w};
      float a1[4] = {rx[r][1].x, rx[r][1].y, rx[r][1].z, rx[r][1].w};
#pragma unroll
      for (int i = 0; i < 4; ++i)
        *(unsigned*)&lds_x[bi][(xnn + i) * 32 + kb] = pack2bf(a0[i], a1[i]);
    }
  };

  v8f acc[4] = {};

  auto computeT = [&](int bi) {
    Frag fb;  // B 32x16: lane = n (lane&15), lanes 16-31 hold K=16..31
    {
      int o = ((wave << 4) + (lane & 15)) * 32 + (lane & 16);
      fb.q[0] = *(const uint4*)&lds_x[bi][o];
      fb.q[1] = *(const uint4*)&lds_x[bi][o + 8];
    }
    Frag fa[4];
#pragma unroll
    for (int t = 0; t < 4; ++t) {
      int o = ((t << 4) + (lane & 15)) * 32 + (lane & 16);
      fa[t].q[0] = *(const uint4*)&lds_a[bi][o];
      fa[t].q[1] = *(const uint4*)&lds_a[bi][o + 8];
    }
#pragma unroll
    for (int t = 0; t < 4; ++t)
      acc[t] = __builtin_amdgcn_wmma_f32_16x16x32_bf16(
          false, fa[t].v, false, fb.v, (short)0, acc[t], false, false);
  };

  loadG(0);
  int bi = 0;
  for (int ko = 0; ko < K; ko += 32) {
    storeL(bi);
    if (ko + 32 < K) loadG(ko + 32);   // overlap next global fetch with WMMAs
    __syncthreads();
    computeT(bi);
    bi ^= 1;
  }

  const int n  = n0 + (wave << 4) + (lane & 15);
  const int mo = (lane & 16) >> 1;
#pragma unroll
  for (int t = 0; t < 4; ++t) {
#pragma unroll
    for (int r = 0; r < 8; ++r) {
      int   mloc = (t << 4) + r + mo;
      int   m    = m0 + mloc;
      float v    = acc[t][r] + bias[m];
      size_t oi  = ((size_t)b * M + m) * L + n;
      if (resid) v += resid[oi];
      Y[oi] = v;
      if (stats) {
        float s = v, q = v * v;
#pragma unroll
        for (int off = 1; off < 16; off <<= 1) {
          s += __shfl_xor(s, off, 32);
          q += __shfl_xor(q, off, 32);
        }
        if ((lane & 15) == 0) {
          atomicAdd(&sdsum[mloc], s);
          atomicAdd(&sdsq[mloc], q);
        }
      }
    }
  }
  if (stats) {
    __syncthreads();
    if (tid < 64) {
      atomicAdd(&stats[m0 + tid],     sdsum[tid]);
      atomicAdd(&stats[M + m0 + tid], sdsq[tid]);
    }
  }
}

// ---------------------------------------------------------------------------
// 64-channel k=3 conv via WMMA im2col (K = 3*64 = 192), pad (1,1).
// Whole weight (64x192 bf16) in LDS; input tile 130 x 64 with halo, [l][c].
// ---------------------------------------------------------------------------
__global__ __launch_bounds__(256) void conv3_bf16(
    const unsigned short* __restrict__ Wg,   // [64,192] bf16 (k = t*64 + c)
    const float* __restrict__ inp, size_t in_bstride,
    const float* __restrict__ bias,          // [64]
    float* __restrict__ Y,                   // [B,64,L]
    float* __restrict__ stats)               // [128]
{
  __shared__ __align__(16) unsigned short lds_w[64 * 192];
  __shared__ __align__(16) unsigned short lds_in[130 * 64];
  __shared__ float sdsum[64];
  __shared__ float sdsq[64];

  const int b    = blockIdx.y;
  const int n0   = blockIdx.x * 128;
  const int tid  = threadIdx.x;
  const int lane = tid & 31;
  const int wave = tid >> 5;

  if (tid < 64) { sdsum[tid] = 0.f; sdsq[tid] = 0.f; }

  for (int idx = tid << 3; idx < 64 * 192; idx += 256 * 8)
    *(uint4*)&lds_w[idx] = *(const uint4*)&Wg[idx];

  // input tile: pack channel pairs -> ds_store_b32, global reads coalesced on l
  const float* xb = inp + (size_t)b * in_bstride;
  for (int idx = tid; idx < 32 * 130; idx += 256) {
    int cp2 = idx / 130;               // channel pair (0..31)
    int ll  = idx - cp2 * 130;         // 0..129
    int c   = cp2 << 1;
    int l   = n0 - 1 + ll;
    float f0 = 0.f, f1 = 0.f;
    if (l >= 0 && l < L_) {
      f0 = xb[(size_t)c * L_ + l];
      f1 = xb[(size_t)(c + 1) * L_ + l];
    }
    *(unsigned*)&lds_in[ll * 64 + c] = pack2bf(f0, f1);
  }
  __syncthreads();

  v8f acc[4] = {};
#pragma unroll
  for (int ks = 0; ks < 192; ks += 32) {
    int t  = ks >> 6;                 // tap index (0..2)
    int cb = (ks & 63) + (lane & 16); // channel base within the tap
    Frag fb;
    int lrow = ((wave << 4) + (lane & 15) + t) * 64 + cb;  // l-(n0-1) = nloc+t
    fb.q[0] = *(const uint4*)&lds_in[lrow];
    fb.q[1] = *(const uint4*)&lds_in[lrow + 8];
    Frag fa[4];
#pragma unroll
    for (int tm = 0; tm < 4; ++tm) {
      int o = ((tm << 4) + (lane & 15)) * 192 + ks + (lane & 16);
      fa[tm].q[0] = *(const uint4*)&lds_w[o];
      fa[tm].q[1] = *(const uint4*)&lds_w[o + 8];
    }
#pragma unroll
    for (int tm = 0; tm < 4; ++tm)
      acc[tm] = __builtin_amdgcn_wmma_f32_16x16x32_bf16(
          false, fa[tm].v, false, fb.v, (short)0, acc[tm], false, false);
  }

  const int n  = n0 + (wave << 4) + (lane & 15);
  const int mo = (lane & 16) >> 1;
#pragma unroll
  for (int t = 0; t < 4; ++t) {
#pragma unroll
    for (int r = 0; r < 8; ++r) {
      int   mloc = (t << 4) + r + mo;
      float v    = acc[t][r] + bias[mloc];
      Y[((size_t)b * 64 + mloc) * L_ + n] = v;
      float s = v, q = v * v;
#pragma unroll
      for (int off = 1; off < 16; off <<= 1) {
        s += __shfl_xor(s, off, 32);
        q += __shfl_xor(q, off, 32);
      }
      if ((lane & 15) == 0) {
        atomicAdd(&sdsum[mloc], s);
        atomicAdd(&sdsq[mloc], q);
      }
    }
  }
  __syncthreads();
  if (tid < 64) {
    atomicAdd(&stats[tid],      sdsum[tid]);
    atomicAdd(&stats[64 + tid], sdsq[tid]);
  }
}

// ---------------------------------------------------------------------------
// DWT low-pass per (b,c) row: symmetric pad, stride-2 analysis, dilated
// synthesis. Also emits high = x - low and the per-row mean of high.
// ---------------------------------------------------------------------------
__global__ __launch_bounds__(128) void dwt_low(
    const float* __restrict__ cs, float* __restrict__ low,
    float* __restrict__ high, float* __restrict__ hmean)
{
  __shared__ float sx[512];
  __shared__ float sca[260];
  __shared__ float red[128];
  const size_t row = blockIdx.x;
  const float* src = cs + row * L_;
  const int tid = threadIdx.x;
#pragma unroll
  for (int i = 0; i < 4; ++i) sx[tid + i * 128] = src[tid + i * 128];
  __syncthreads();
  for (int k = tid; k < 259; k += 128) {
    float s = 0.f;
#pragma unroll
    for (int i = 0; i < 8; ++i) {
      int q  = 2 * k + i - 6;                        // symmetric-extended index
      int qq = (q < 0) ? (-1 - q) : ((q < 512) ? q : (1023 - q));
      s += c_rec_lo[i] * sx[qq];
    }
    sca[k] = s;
  }
  __syncthreads();
  float hsum = 0.f;
#pragma unroll
  for (int i = 0; i < 4; ++i) {
    int nn = tid + i * 128;
    float s = 0.f;
#pragma unroll
    for (int j = 0; j < 8; ++j) {
      int m = nn + j - 1;                            // dilated, pad 1
      if (m >= 0 && (m & 1) == 0) s += c_rec_lo[7 - j] * sca[m >> 1];
    }
    float h = sx[nn] - s;
    low [row * L_ + nn] = s;
    high[row * L_ + nn] = h;
    hsum += h;
  }
  red[tid] = hsum;
  __syncthreads();
  for (int s = 64; s > 0; s >>= 1) {
    if (tid < s) red[tid] += red[tid + s];
    __syncthreads();
  }
  if (tid == 0) hmean[row] = red[0] * (1.f / 512.f);
}

// ---------------------------------------------------------------------------
__global__ void bn_finalize(const float* __restrict__ stats,
                            float* __restrict__ meanrs, int M, float invCnt)
{
  int m = blockIdx.x * blockDim.x + threadIdx.x;
  if (m < M) {
    float mean = stats[m] * invCnt;
    float var  = stats[M + m] * invCnt - mean * mean;
    meanrs[m]     = mean;
    meanrs[M + m] = rsqrtf(var + 1e-5f);
  }
}

// block BN+ReLU; writes into concat buffer slice and builds next block input
__global__ __launch_bounds__(256) void blk_apply(
    const float* __restrict__ y, const float* __restrict__ meanrs,
    const float* __restrict__ g, const float* __restrict__ beta,
    float* __restrict__ cs_out, size_t cs_bstride,
    const float* __restrict__ feats_next, size_t fn_bstride,
    float* __restrict__ inp_next, size_t in_bstride)
{
  size_t rc = blockIdx.x;            // b*64 + c
  int    c  = (int)(rc & 63);
  size_t b  = rc >> 6;
  float mean = meanrs[c], rs = meanrs[64 + c];
  float gg = g[c], bb = beta[c];
  const float* yr = y + rc * L_;
  float* co = cs_out + b * cs_bstride + (size_t)c * L_;
  for (int l = threadIdx.x; l < L_; l += 256) {
    float v = fmaxf((yr[l] - mean) * rs * gg + bb, 0.f);
    co[l] = v;
    if (feats_next)
      inp_next[b * in_bstride + (size_t)c * L_ + l] =
          feats_next[b * fn_bstride + (size_t)c * L_ + l] + v;
  }
}

// relu(bn(y)) optionally multiplied by `mul` (gating) -> out
__global__ __launch_bounds__(256) void bn_apply_ew(
    const float* __restrict__ y, const float* __restrict__ meanrs,
    const float* __restrict__ g, const float* __restrict__ beta,
    const float* __restrict__ mul, float* __restrict__ out, int M)
{
  size_t total = (size_t)B_ * M * L_;
  for (size_t i = (size_t)blockIdx.x * 256 + threadIdx.x; i < total;
       i += (size_t)gridDim.x * 256) {
    int   c = (int)((i / L_) % M);
    float v = fmaxf((y[i] - meanrs[c]) * meanrs[M + c] * g[c] + beta[c], 0.f);
    out[i] = mul ? mul[i] * v : v;
  }
}

// lc[b,o] = sigmoid( W_l2g[o,:] . hmean[b,:] + b_l2g[o] )
__global__ __launch_bounds__(256) void lc_sigmoid(
    const float* __restrict__ hmean, const float* __restrict__ W,
    const float* __restrict__ bias, float* __restrict__ lc)
{
  __shared__ float sh[C_];
  int b = blockIdx.x, o = threadIdx.x;
  sh[o] = hmean[(size_t)b * C_ + o];
  __syncthreads();
  float s = bias[o];
  const float* wr = W + (size_t)o * C_;
  for (int c = 0; c < C_; ++c) s += wr[c] * sh[c];
  lc[(size_t)b * C_ + o] = 1.f / (1.f + __expf(-s));
}

// enh_long: buf[b,c,l] *= lc[b,c]
__global__ __launch_bounds__(256) void scale_rows(
    float* __restrict__ buf, const float* __restrict__ lc)
{
  size_t total = (size_t)B_ * C_ * L_;
  for (size_t i = (size_t)blockIdx.x * 256 + threadIdx.x; i < total;
       i += (size_t)gridDim.x * 256)
    buf[i] *= lc[i / L_];
}

__global__ void cvt_mat(const float* __restrict__ src,
                        unsigned short* __restrict__ dst, int n)
{
  for (int i = blockIdx.x * 256 + threadIdx.x; i < n; i += gridDim.x * 256)
    dst[i] = f2bfu(src[i]);
}

// Wb[i,o,c,t] -> dst[i][o][k = t*64 + c] bf16
__global__ void cvt_convw(const float* __restrict__ Wb,
                          unsigned short* __restrict__ dst)
{
  int total = NB_ * BC_ * 192;
  for (int idx = blockIdx.x * 256 + threadIdx.x; idx < total;
       idx += gridDim.x * 256) {
    int i   = idx / (BC_ * 192);
    int rem = idx - i * (BC_ * 192);
    int o   = rem / 192;
    int k   = rem - o * 192;
    int t   = k >> 6;
    int c   = k & 63;
    dst[idx] = f2bfu(Wb[(((size_t)i * BC_ + o) * BC_ + c) * 3 + t]);
  }
}

__global__ void zero_f(float* p, int n)
{
  int i = blockIdx.x * 256 + threadIdx.x;
  if (i < n) p[i] = 0.f;
}

// ---------------------------------------------------------------------------
extern "C" void kernel_launch(void* const* d_in, const int* in_sizes, int n_in,
                              void* d_out, int out_size, void* d_ws, size_t ws_size,
                              hipStream_t stream)
{
  (void)in_sizes; (void)n_in; (void)out_size; (void)ws_size;
  const float* x        = (const float*)d_in[0];
  const float* W_init   = (const float*)d_in[1];
  const float* b_init   = (const float*)d_in[2];
  const float* Wb       = (const float*)d_in[3];
  const float* bb       = (const float*)d_in[4];
  const float* gb       = (const float*)d_in[5];
  const float* betab    = (const float*)d_in[6];
  const float* W_g2l    = (const float*)d_in[7];
  const float* b_g2l    = (const float*)d_in[8];
  const float* g_g2l    = (const float*)d_in[9];
  const float* beta_g2l = (const float*)d_in[10];
  const float* W_l2g    = (const float*)d_in[11];
  const float* b_l2g    = (const float*)d_in[12];
  const float* Wf       = (const float*)d_in[13];
  const float* bf       = (const float*)d_in[14];
  const float* gf       = (const float*)d_in[15];
  const float* betaf    = (const float*)d_in[16];
  const float* W_fuse   = (const float*)d_in[17];
  const float* b_fuse   = (const float*)d_in[18];

  char* wsb = (char*)d_ws;
  size_t off = 0;
  auto alloc = [&](size_t bytes) {
    void* p = wsb + off;
    off = (off + bytes + 255) & ~(size_t)255;
    return p;
  };
  unsigned short* wgA = (unsigned short*)alloc((size_t)C_ * C_ * 2);
  unsigned short* wgG = (unsigned short*)alloc((size_t)C_ * C_ * 2);
  unsigned short* wgF = (unsigned short*)alloc((size_t)C_ * 2 * C_ * 2);
  unsigned short* wgU = (unsigned short*)alloc((size_t)C_ * C_ * 2);
  unsigned short* wgB = (unsigned short*)alloc((size_t)NB_ * BC_ * 192 * 2);
  float* statsB = (float*)alloc(NB_ * 128 * 4);   // 4 blocks x (sum64|sq64)
  float* statsG = (float*)alloc(512 * 4);
  float* statsF = (float*)alloc(512 * 4);
  float* mrsB   = (float*)alloc(NB_ * 128 * 4);
  float* mrsG   = (float*)alloc(512 * 4);
  float* mrsF   = (float*)alloc(512 * 4);
  float* hmean  = (float*)alloc((size_t)B_ * C_ * 4);
  float* lc     = (float*)alloc((size_t)B_ * C_ * 4);
  const size_t big = (size_t)B_ * C_ * L_ * 4;
  float* feats = (float*)alloc(big);               // later dead
  float* csb   = (float*)alloc(big);               // concat -> fused (reused)
  float* lowb  = (float*)alloc(big);               // low -> enh_long (in place)
  float* highb = (float*)alloc(big);               // high -> enh_short (in place)
  float* y2b   = (float*)alloc(big);               // y2 -> y3 (reused)
  float* inpb  = (float*)alloc((size_t)B_ * BC_ * L_ * 4);
  float* yblk  = (float*)alloc((size_t)B_ * BC_ * L_ * 4);

  const float invBL = 1.f / ((float)B_ * (float)L_);
  const dim3 tpb(256);
  const dim3 ggrid(L_ / 128, C_ / 64, B_);

  // stats are contiguous (each chunk is 2048 B, alloc align 256) -> one zero
  zero_f<<<dim3(6), tpb, 0, stream>>>(statsB, NB_ * 128 + 512 + 512);

  // weight conversions fp32 -> bf16
  cvt_mat<<<dim3(64),  tpb, 0, stream>>>(W_init, wgA, C_ * C_);
  cvt_mat<<<dim3(64),  tpb, 0, stream>>>(W_g2l,  wgG, C_ * C_);
  cvt_mat<<<dim3(128), tpb, 0, stream>>>(Wf,     wgF, C_ * 2 * C_);
  cvt_mat<<<dim3(64),  tpb, 0, stream>>>(W_fuse, wgU, C_ * C_);
  cvt_convw<<<dim3(48), tpb, 0, stream>>>(Wb, wgB);

  // feats = W_init . x + b_init
  gemm_bf16<<<ggrid, tpb, 0, stream>>>(
      wgA, x, x, b_init, (const float*)nullptr, feats, (float*)nullptr,
      C_, C_, L_, C_);

  // dense block chain (conv3 -> BN stats -> BN+ReLU, fused next-input build)
  for (int i = 0; i < NB_; ++i) {
    const float* cin = (i == 0) ? feats : inpb;
    size_t cbs = (i == 0) ? (size_t)C_ * L_ : (size_t)BC_ * L_;
    conv3_bf16<<<dim3(L_ / 128, B_), tpb, 0, stream>>>(
        wgB + (size_t)i * BC_ * 192, cin, cbs, bb + i * BC_, yblk,
        statsB + i * 128);
    bn_finalize<<<dim3(1), dim3(64), 0, stream>>>(
        statsB + i * 128, mrsB + i * 128, BC_, invBL);
    blk_apply<<<dim3(B_ * BC_), tpb, 0, stream>>>(
        yblk, mrsB + i * 128, gb + i * BC_, betab + i * BC_,
        csb + (size_t)i * BC_ * L_, (size_t)C_ * L_,
        (i < NB_ - 1) ? (const float*)(feats + (size_t)(i + 1) * BC_ * L_)
                      : (const float*)nullptr,
        (size_t)C_ * L_, inpb, (size_t)BC_ * L_);
  }

  // DWT split: low / high / per-row mean of high
  dwt_low<<<dim3(B_ * C_), dim3(128), 0, stream>>>(csb, lowb, highb, hmean);

  // gc path: y2 = W_g2l . low + b ; BN ; enh_short = high * relu(bn(y2))
  gemm_bf16<<<ggrid, tpb, 0, stream>>>(
      wgG, lowb, lowb, b_g2l, (const float*)nullptr, y2b, statsG,
      C_, C_, L_, C_);
  bn_finalize<<<dim3(1), tpb, 0, stream>>>(statsG, mrsG, C_, invBL);
  bn_apply_ew<<<dim3(4096), tpb, 0, stream>>>(
      y2b, mrsG, g_g2l, beta_g2l, highb, highb, C_);

  // lc path: enh_long = low * sigmoid(W_l2g . mean(high))
  lc_sigmoid<<<dim3(B_), dim3(C_), 0, stream>>>(hmean, W_l2g, b_l2g, lc);
  scale_rows<<<dim3(4096), tpb, 0, stream>>>(lowb, lc);

  // fuse: y3 = Wf . cat(enh_long, enh_short) + bf ; fused = relu(bn(y3))
  gemm_bf16<<<ggrid, tpb, 0, stream>>>(
      wgF, lowb, highb, bf, (const float*)nullptr, y2b, statsF,
      C_, 2 * C_, L_, C_);
  bn_finalize<<<dim3(1), tpb, 0, stream>>>(statsF, mrsF, C_, invBL);
  bn_apply_ew<<<dim3(4096), tpb, 0, stream>>>(
      y2b, mrsF, gf, betaf, (const float*)nullptr, csb, C_);

  // out = W_fuse . fused + b_fuse + x
  gemm_bf16<<<ggrid, tpb, 0, stream>>>(
      wgU, csb, csb, b_fuse, x, (float*)d_out, (float*)nullptr,
      C_, C_, L_, C_);
}
